// truncated_krylov_layer_73280732004595
// MI455X (gfx1250) — compile-verified, matrix operand order checked by reading, not run on hardware
//
#include <hip/hip_runtime.h>
#include <hip/hip_bf16.h>
#include <stdint.h>

// =====================================================================
// Truncated Krylov GCN layer for gfx1250 (MI455X)
//  feats = [X, AX, A^2X, A^3X];  out = concat(feats) @ W + b
//  - SPMM: wave-per-edge, coalesced f32 atomics (X is L2-resident: 102MB < 192MB L2)
//  - GEMM: bf16 split-precision (hi/lo) via v_wmma_f32_16x16x32_bf16.
//    One 512-thread block per 16-row M-panel: the f32->bf16(hi/lo) split of
//    the A panel is done ONCE into LDS (amortized across all 16 n-tiles),
//    so the WMMA loop is ds_load_b128 + global b128 + v_wmma only.
// Workspace layout (needs ~308.2 MB):
//   [0)              f1 = A X        (N*256 f32)
//   [1*N*256 f32)    f2 = A^2 X
//   [2*N*256 f32)    f3 = A^3 X
//   [3*N*256 f32)    W fragments: 512 entries * 512 dwords (hi+lo) = 1 MB
// =====================================================================

typedef __attribute__((ext_vector_type(16))) __bf16 v16bf;
typedef __attribute__((ext_vector_type(8)))  float  v8f;

union Frag16 { v16bf v; uint32_t u[8]; };

// Split two consecutive f32 (even-k, odd-k) into packed-bf16 hi and lo dwords.
// hi = truncated top-16 bits (1x v_perm_b32); lo = residual rounded half-up.
// Combined (hi + lo) representation error ~2^-17 relative.
__device__ __forceinline__ void split_pair(float e0, float e1,
                                           uint32_t& hi, uint32_t& lo) {
  uint32_t b0 = __float_as_uint(e0);
  uint32_t b1 = __float_as_uint(e1);
  hi = __builtin_amdgcn_perm(b1, b0, 0x07060302u);   // {b1[31:16], b0[31:16]}
  float h0 = __uint_as_float(b0 & 0xffff0000u);
  float h1 = __uint_as_float(b1 & 0xffff0000u);
  uint32_t r0 = __float_as_uint(e0 - h0) + 0x8000u;  // round residual to nearest
  uint32_t r1 = __float_as_uint(e1 - h1) + 0x8000u;
  lo = __builtin_amdgcn_perm(r1, r0, 0x07060302u);
}

// Split 4 float4's (16 elements, k rel kbase+0..7 and kbase+16..23, in A/B fragment order) into
// 8 hi dwords + 8 lo dwords.
__device__ __forceinline__ void split8(const float4& x0, const float4& x1,
                                       const float4& x2, const float4& x3,
                                       uint32_t* hi, uint32_t* lo) {
  split_pair(x0.x, x0.y, hi[0], lo[0]);
  split_pair(x0.z, x0.w, hi[1], lo[1]);
  split_pair(x1.x, x1.y, hi[2], lo[2]);
  split_pair(x1.z, x1.w, hi[3], lo[3]);
  split_pair(x2.x, x2.y, hi[4], lo[4]);
  split_pair(x2.z, x2.w, hi[5], lo[5]);
  split_pair(x3.x, x3.y, hi[6], lo[6]);
  split_pair(x3.z, x3.w, hi[7], lo[7]);
}

// ---------------------------------------------------------------------
// SPMM: one wave32 per edge; 8 fully coalesced loads + 8 coalesced
// f32 atomic-adds per lane group (256 features).
// ---------------------------------------------------------------------
__global__ void spmm_kernel(const int* __restrict__ esrc,
                            const int* __restrict__ edst,
                            const float* __restrict__ ew,
                            const float* __restrict__ fin,
                            float* __restrict__ fout, int E) {
  int gid  = blockIdx.x * blockDim.x + threadIdx.x;
  int e    = gid >> 5;
  int lane = gid & 31;
  if (e >= E) return;

  if (lane == 0) {
    int pe = e + 4096; if (pe >= E) pe = E - 1;
    __builtin_prefetch(esrc + pe, 0, 0);   // global_prefetch_b8
    __builtin_prefetch(edst + pe, 0, 0);
    __builtin_prefetch(ew   + pe, 0, 0);
  }

  int   s  = esrc[e];
  int   d  = edst[e];
  float wt = ew[e];
  const float* in  = fin  + (size_t)s * 256;
  float*       out = fout + (size_t)d * 256;
#pragma unroll
  for (int j = 0; j < 8; ++j) {
    int c = j * 32 + lane;
    atomicAdd(out + c, wt * in[c]);        // global_atomic_add_f32, coalesced
  }
}

// ---------------------------------------------------------------------
// Pre-pack shared_weight [1024,256] into wave32 WMMA B-fragment layout,
// bf16 hi/lo split. Entry = (blk, n_tile, kk) -> 512 dwords
// (256 hi then 256 lo), dword index within half = lane*8 + vgpr_i.
// B element for lane L, vgpr i: n = ntile*16 + (L&15),
//   kbase = (L<16)?0:8, krel = i<4 ? kbase+2i : 16+kbase+2(i-4)
// ---------------------------------------------------------------------
__global__ void wprep_kernel(const float* __restrict__ W,
                             uint32_t* __restrict__ frag) {
  int gid   = blockIdx.x * blockDim.x + threadIdx.x;  // 0..16383
  int lane  = gid & 31;
  int entry = gid >> 5;                               // 0..511
  if (entry >= 512) return;
  int kk  = entry & 7;
  int nt  = (entry >> 3) & 15;
  int blk = entry >> 7;
  int n     = nt * 16 + (lane & 15);
  int kbase = (lane < 16) ? 0 : 8;
  uint32_t* base = frag + (size_t)entry * 512 + lane * 8;
#pragma unroll
  for (int i = 0; i < 8; ++i) {
    int krel = (i < 4) ? (kbase + 2 * i) : (16 + kbase + 2 * (i - 4));
    int k    = blk * 256 + kk * 32 + krel;
    float w0 = W[(size_t)k * 256 + n];
    float w1 = W[(size_t)(k + 1) * 256 + n];
    uint32_t hi, lo;
    split_pair(w0, w1, hi, lo);
    base[i]       = hi;
    base[256 + i] = lo;
  }
}

// ---------------------------------------------------------------------
// GEMM: one 512-thread block (16 waves) per 16-row M-panel.
// Phase 1: cooperatively split the 16x1024 f32 panel into bf16 hi/lo
//          WMMA A-fragments in LDS (64 KB).
// Phase 2: wave w computes n-tile w; per K=32 step:
//          2x ds_load_b128 (ahi) + 2x (alo)  [broadcast across waves]
//          2x global b128 (bhi) + 2x (blo)   [1 MB buffer, L2-resident]
//          3x v_wmma_f32_16x16x32_bf16 (hi*hi + hi*lo + lo*hi).
// ---------------------------------------------------------------------
__global__ void __launch_bounds__(512) gemm_kernel(
    const float* __restrict__ x,
    const float* __restrict__ f1,
    const float* __restrict__ f2,
    const float* __restrict__ f3,
    const uint32_t* __restrict__ wfrag,
    const float* __restrict__ bias,
    float* __restrict__ out,
    int n_mtiles) {
  __shared__ uint32_t smem[32 * 512];   // 64 KB: per kstep 256 hi + 256 lo dwords

  const int t      = threadIdx.x;
  const int m_tile = blockIdx.x;
  if (m_tile >= n_mtiles) return;

  const float* srcs[4] = {x, f1, f2, f3};

  // ---- Phase 1: stage split A panel into LDS fragment layout ----
#pragma unroll
  for (int half = 0; half < 2; ++half) {
    int c     = t + half * 512;          // (kstep, lane) combo: 0..1023
    int ks    = c >> 5;                  // 0..31 (K/32 step over all 4 blocks)
    int L     = c & 31;                  // fragment lane
    int blk   = ks >> 3;
    int kk    = ks & 7;
    int kbase = (L < 16) ? 0 : 8;
    int row   = m_tile * 16 + (L & 15);
    const float* ap = srcs[blk] + (size_t)row * 256 + kk * 32 + kbase;
    float4 x0 = *(const float4*)(ap + 0);     // k rel kbase+0..3
    float4 x1 = *(const float4*)(ap + 4);     // k rel kbase+4..7
    float4 x2 = *(const float4*)(ap + 16);    // k rel kbase+16..19
    float4 x3 = *(const float4*)(ap + 20);    // k rel kbase+20..23
    uint32_t* hi = smem + ks * 512 + L * 8;
    split8(x0, x1, x2, x3, hi, hi + 256);
  }
  __syncthreads();

  // ---- Phase 2: WMMA over K = 1024 ----
  const int lane   = t & 31;
  const int wave   = t >> 5;             // n_tile (0..15)
  v8f acc = {};

#pragma unroll 8
  for (int ks = 0; ks < 32; ++ks) {
    int blk = ks >> 3;
    int kk  = ks & 7;

    const uint4* ah = (const uint4*)(smem + ks * 512 + lane * 8);
    const uint4* al = (const uint4*)(smem + ks * 512 + 256 + lane * 8);
    uint4 a0 = ah[0], a1 = ah[1];
    uint4 b0 = al[0], b1 = al[1];

    const uint32_t* bfr = wfrag +
        (size_t)((blk * 16 + wave) * 8 + kk) * 512 + lane * 8;
    const uint4* bh = (const uint4*)(bfr);
    const uint4* bl = (const uint4*)(bfr + 256);
    uint4 c0 = bh[0], c1 = bh[1];
    uint4 d0 = bl[0], d1 = bl[1];

    Frag16 ahi, alo, bhi, blo;
    ahi.u[0] = a0.x; ahi.u[1] = a0.y; ahi.u[2] = a0.z; ahi.u[3] = a0.w;
    ahi.u[4] = a1.x; ahi.u[5] = a1.y; ahi.u[6] = a1.z; ahi.u[7] = a1.w;
    alo.u[0] = b0.x; alo.u[1] = b0.y; alo.u[2] = b0.z; alo.u[3] = b0.w;
    alo.u[4] = b1.x; alo.u[5] = b1.y; alo.u[6] = b1.z; alo.u[7] = b1.w;
    bhi.u[0] = c0.x; bhi.u[1] = c0.y; bhi.u[2] = c0.z; bhi.u[3] = c0.w;
    bhi.u[4] = c1.x; bhi.u[5] = c1.y; bhi.u[6] = c1.z; bhi.u[7] = c1.w;
    blo.u[0] = d0.x; blo.u[1] = d0.y; blo.u[2] = d0.z; blo.u[3] = d0.w;
    blo.u[4] = d1.x; blo.u[5] = d1.y; blo.u[6] = d1.z; blo.u[7] = d1.w;

    acc = __builtin_amdgcn_wmma_f32_16x16x32_bf16(
        false, ahi.v, false, bhi.v, (short)0, acc, false, false);
    acc = __builtin_amdgcn_wmma_f32_16x16x32_bf16(
        false, ahi.v, false, blo.v, (short)0, acc, false, false);
    acc = __builtin_amdgcn_wmma_f32_16x16x32_bf16(
        false, alo.v, false, bhi.v, (short)0, acc, false, false);
  }

  // C layout: vgpr r, lanes 0-15 -> M=r, lanes 16-31 -> M=8+r; N = lane&15.
  int   n = wave * 16 + (lane & 15);
  float b = bias[n];
  int mbase = m_tile * 16 + ((lane >> 4) << 3);
#pragma unroll
  for (int r = 0; r < 8; ++r)
    out[(size_t)(mbase + r) * 256 + n] = acc[r] + b;
}

// ---------------------------------------------------------------------
extern "C" void kernel_launch(void* const* d_in, const int* in_sizes, int n_in,
                              void* d_out, int out_size, void* d_ws, size_t ws_size,
                              hipStream_t stream) {
  const float* x    = (const float*)d_in[0];
  const int*   esrc = (const int*)  d_in[1];
  const int*   edst = (const int*)  d_in[2];
  const float* ew   = (const float*)d_in[3];
  const float* W    = (const float*)d_in[4];
  const float* bias = (const float*)d_in[5];
  float*       out  = (float*)d_out;

  const int E = in_sizes[1];
  const int N = in_sizes[0] / 256;          // D_IN = 256
  const size_t featElems = (size_t)N * 256;

  float* f1 = (float*)d_ws;
  float* f2 = f1 + featElems;
  float* f3 = f2 + featElems;
  uint32_t* wfrag = (uint32_t*)(f3 + featElems);   // 1 MB

  // Zero the three SPMM accumulation buffers (atomics accumulate).
  hipMemsetAsync(f1, 0, featElems * 3 * sizeof(float), stream);

  // Pre-pack W into bf16 hi/lo WMMA B-fragments (512 entries * 32 lanes).
  wprep_kernel<<<64, 256, 0, stream>>>(W, wfrag);

  // Krylov hops: f1 = A x, f2 = A f1, f3 = A f2.
  int spmmBlocks = (E + 7) / 8;             // 8 waves/block, wave per edge
  spmm_kernel<<<spmmBlocks, 256, 0, stream>>>(esrc, edst, ew, x,  f1, E);
  spmm_kernel<<<spmmBlocks, 256, 0, stream>>>(esrc, edst, ew, f1, f2, E);
  spmm_kernel<<<spmmBlocks, 256, 0, stream>>>(esrc, edst, ew, f2, f3, E);

  // Fused K=1024 GEMM + bias over all four blocks, single HBM pass.
  int n_mtiles = N / 16;                    // 100000/16 = 6250 exactly
  gemm_kernel<<<n_mtiles, 512, 0, stream>>>(x, f1, f2, f3, wfrag, bias, out,
                                            n_mtiles);
}
